// Disentangler_32091995636155
// MI455X (gfx1250) — matrix operand order repaired and checked by reading, not compiled, not run on hardware
//
#include <hip/hip_runtime.h>
#include <hip/hip_bf16.h>
#include <math.h>

typedef __attribute__((ext_vector_type(16))) __bf16 v16bf;
typedef __attribute__((ext_vector_type(8)))  __bf16 v8bf;
typedef __attribute__((ext_vector_type(8)))  float  v8f;
typedef __attribute__((ext_vector_type(4)))  float  f4v;

#define T_DIM      8
#define NTOK       12288
#define D_MODEL    256
#define NNODE      8192
#define NNODES     32768
#define COMP_LEN   16
#define COMP_DIM   64
#define CHUNK      2048      // NNODES / COMP_LEN
#define H1         128       // 2*COMP_DIM
#define LCAP       16        // slot list capacity (P(overflow) ~ 1e-23)

__device__ __forceinline__ float gelu_exact(float v) {
    return 0.5f * v * (1.0f + erff(v * 0.70710678118654752f));
}

// ---------------------------------------------------------------------------
// Pack a row-major fp32 weight (K x N) into per-lane WMMA bf16 B-frag layout:
// frag (kt,nt) -> 32 lanes -> 16 bf16 elems; lane: N = nt*16 + (lane&15),
// K = kt*32 + 16*(lane>>4) + e.  One 32B aligned load per frag per lane later.
// ---------------------------------------------------------------------------
__global__ __launch_bounds__(256)
void pack_b_kernel(const float* __restrict__ w, __bf16* __restrict__ wp,
                   int K, int N) {
    int tid = blockIdx.x * 256 + threadIdx.x;
    if (tid >= K * N) return;
    int e    = tid & 15;
    int lane = (tid >> 4) & 31;
    int frag = tid >> 9;              // 512 elems per frag
    int ntiles = N >> 4;
    int kt = frag / ntiles;
    int nt = frag - kt * ntiles;
    int Krow = kt * 32 + 16 * (lane >> 4) + e;
    int Ncol = nt * 16 + (lane & 15);
    wp[tid] = (__bf16)w[Krow * N + Ncol];
}

// ---------------------------------------------------------------------------
// Invert the scatter: per (t, slot) token lists.  One ticket atomic per token
// (65536 total, vs 16.7M value atomics in a materialized-tee design).
// ---------------------------------------------------------------------------
__global__ __launch_bounds__(256)
void count_fill_kernel(const int* __restrict__ indices,
                       int* __restrict__ cnt, unsigned short* __restrict__ list) {
    int tid = blockIdx.x * 256 + threadIdx.x;     // 0 .. T_DIM*NNODE-1
    int t = tid >> 13;
    int i = tid & (NNODE - 1);
    int slot = indices[tid];
    int pos = atomicAdd(&cnt[t * NNODES + slot], 1);
    if (pos < LCAP) list[((size_t)(t * NNODES + slot)) * LCAP + pos] = (unsigned short)i;
}

// ---------------------------------------------------------------------------
// Fused gather-encoder.  Per wave: one 16-slot tile.
//   phase 1: for each slot row, gather its tokens' x rows (wave-cooperative,
//            NT loads), LayerNorm(256) inline via shfl reduction, accumulate
//            fp32 row into LDS.  (each token consumed exactly once)
//   phase 2: GEMM1 (K=256,N=128) v_wmma_f32_16x16x32_bf16, +b1, exact gelu,
//            bf16 transpose via LDS, GEMM2 (K=128,N=64), reduce 16 rows,
//            atomicAdd into comp[t, chunk, 64] (+16*b2 per tile).
// Block = 2 waves -> 40KB static LDS (under the 64KB static cap).
// ---------------------------------------------------------------------------
__global__ __launch_bounds__(64)
void encoder_kernel(const float* __restrict__ x,
                    const int* __restrict__ cnt, const unsigned short* __restrict__ list,
                    const float* __restrict__ ln_g, const float* __restrict__ ln_b,
                    const v16bf* __restrict__ w1p, const float* __restrict__ b1,
                    const v16bf* __restrict__ w2p, const float* __restrict__ b2,
                    float* __restrict__ comp) {
    __shared__ __align__(16) float  ars[2 * 16 * D_MODEL];  // 32 KB: fp32 A tiles
    __shared__ __align__(16) __bf16 hbuf[2 * 16 * H1];      //  8 KB: bf16 h tiles

    const int w    = threadIdx.x >> 5;
    const int lane = threadIdx.x & 31;
    const int half = lane >> 4;
    const int m16  = lane & 15;

    const int gt   = blockIdx.x * 2 + w;   // global 16-slot tile id (0..16383)
    const int t    = gt >> 11;             // 2048 tiles per t
    const int mt   = gt & 2047;
    const int row0 = mt << 4;
    const int l    = mt >> 7;              // 128 tiles per 2048-slot chunk

    // ---- phase 1: gather + inline LayerNorm into LDS row tile -------------
    const int col0 = lane * 8;
    float gg[8], bb[8];
#pragma unroll
    for (int j = 0; j < 8; ++j) { gg[j] = ln_g[col0 + j]; bb[j] = ln_b[col0 + j]; }

    float* arow = ars + w * (16 * D_MODEL);
    const float* xbase = x + (size_t)t * NTOK * D_MODEL;

    for (int r = 0; r < 16; ++r) {
        const int slot = row0 + r;
        int c = cnt[t * NNODES + slot];
        if (c > LCAP) c = LCAP;
        const unsigned short* lst = list + ((size_t)(t * NNODES + slot)) * LCAP;
        float acc[8] = {0.f, 0.f, 0.f, 0.f, 0.f, 0.f, 0.f, 0.f};
        for (int k = 0; k < c; ++k) {
            const int tok = lst[k];                       // wave-uniform
            const float* xr = xbase + (size_t)tok * D_MODEL + col0;
            f4v v0 = __builtin_nontemporal_load((const f4v*)xr);
            f4v v1 = __builtin_nontemporal_load((const f4v*)(xr + 4));
            float in[8] = {v0.x, v0.y, v0.z, v0.w, v1.x, v1.y, v1.z, v1.w};
            float s = 0.0f, ss = 0.0f;
#pragma unroll
            for (int j = 0; j < 8; ++j) { s += in[j]; ss += in[j] * in[j]; }
#pragma unroll
            for (int m = 16; m >= 1; m >>= 1) {
                s  += __shfl_xor(s,  m, 32);
                ss += __shfl_xor(ss, m, 32);
            }
            const float mean = s * (1.0f / 256.0f);
            const float var  = ss * (1.0f / 256.0f) - mean * mean;
            const float inv  = rsqrtf(var + 1e-5f);
#pragma unroll
            for (int j = 0; j < 8; ++j)
                acc[j] += (in[j] - mean) * inv * gg[j] + bb[j];
        }
        *(f4v*)(arow + r * D_MODEL + col0)     = *(f4v*)&acc[0];
        *(f4v*)(arow + r * D_MODEL + col0 + 4) = *(f4v*)&acc[4];
    }

    // ---- phase 2: GEMM1 -> gelu -> GEMM2 -> chunk reduce ------------------
    const float* amine = arow + m16 * D_MODEL;   // this lane's A row (M = lane&15)
    v8f c1[8] = {};
#pragma unroll
    for (int kt = 0; kt < 8; ++kt) {
        const int k0 = kt * 32 + half * 16;
        f4v f0 = *(const f4v*)(amine + k0);
        f4v f1 = *(const f4v*)(amine + k0 + 4);
        f4v f2 = *(const f4v*)(amine + k0 + 8);
        f4v f3 = *(const f4v*)(amine + k0 + 12);
        v16bf a;
        a[0]=(__bf16)f0.x; a[1]=(__bf16)f0.y; a[2]=(__bf16)f0.z; a[3]=(__bf16)f0.w;
        a[4]=(__bf16)f1.x; a[5]=(__bf16)f1.y; a[6]=(__bf16)f1.z; a[7]=(__bf16)f1.w;
        a[8]=(__bf16)f2.x; a[9]=(__bf16)f2.y; a[10]=(__bf16)f2.z; a[11]=(__bf16)f2.w;
        a[12]=(__bf16)f3.x; a[13]=(__bf16)f3.y; a[14]=(__bf16)f3.z; a[15]=(__bf16)f3.w;
#pragma unroll
        for (int nt = 0; nt < 8; ++nt) {
            v16bf bf = w1p[(kt * 8 + nt) * 32 + lane];
            c1[nt] = __builtin_amdgcn_wmma_f32_16x16x32_bf16(
                false, a, false, bf, (short)0, c1[nt], false, false);
        }
    }

    // bias + exact gelu; store 16x128 tile (bf16) to this wave's LDS slab
    __bf16* hb = hbuf + w * (16 * H1);
#pragma unroll
    for (int nt = 0; nt < 8; ++nt) {
        const int n = nt * 16 + m16;
        const float bias = b1[n];
#pragma unroll
        for (int r = 0; r < 8; ++r) {
            float v = c1[nt][r] + bias;            // C layout: M = r + 8*half
            hb[(r + half * 8) * H1 + n] = (__bf16)gelu_exact(v);
        }
    }
    __syncthreads();

    v8f c2[4] = {};
    const __bf16* hrow = hbuf + w * (16 * H1) + m16 * H1;
#pragma unroll
    for (int kt = 0; kt < 4; ++kt) {
        const int k0 = kt * 32 + half * 16;
        v8bf lo = *(const v8bf*)(hrow + k0);
        v8bf hi = *(const v8bf*)(hrow + k0 + 8);
        v16bf a;
#pragma unroll
        for (int j = 0; j < 8; ++j) { a[j] = lo[j]; a[8 + j] = hi[j]; }
#pragma unroll
        for (int nt = 0; nt < 4; ++nt) {
            v16bf bf = w2p[(kt * 4 + nt) * 32 + lane];
            c2[nt] = __builtin_amdgcn_wmma_f32_16x16x32_bf16(
                false, a, false, bf, (short)0, c2[nt], false, false);
        }
    }

    // reduce the 16 rows of this tile into comp[t, l, :]; add 16*b2 per tile
    float* crow = comp + ((size_t)t * COMP_LEN + l) * COMP_DIM;
#pragma unroll
    for (int nt = 0; nt < 4; ++nt) {
        float s = 0.0f;
#pragma unroll
        for (int r = 0; r < 8; ++r) s += c2[nt][r];
        s += __shfl_xor(s, 16, 32);
        if (lane < 16)
            atomicAdd(crow + nt * 16 + lane, s + 16.0f * b2[nt * 16 + lane]);
    }
}

// ---------------------------------------------------------------------------
// comp (8,16,64): LN over 1024 (lnf), then per-l LN over 64 (lnd) -> tee2d.
// One wave per t.
// ---------------------------------------------------------------------------
__global__ __launch_bounds__(32)
void comp_ln_kernel(const float* __restrict__ comp,
                    const float* __restrict__ lnf_g, const float* __restrict__ lnf_b,
                    const float* __restrict__ lnd_g, const float* __restrict__ lnd_b,
                    float* __restrict__ tee2d) {
    __shared__ float sh[1024];
    const int t = blockIdx.x, lane = threadIdx.x;

    float vals[32];
    float s = 0.0f, ss = 0.0f;
#pragma unroll
    for (int k = 0; k < 32; ++k) {
        float v = comp[t * 1024 + k * 32 + lane];
        vals[k] = v; s += v; ss += v * v;
    }
#pragma unroll
    for (int m = 16; m >= 1; m >>= 1) {
        s  += __shfl_xor(s,  m, 32);
        ss += __shfl_xor(ss, m, 32);
    }
    const float mean = s * (1.0f / 1024.0f);
    const float var  = ss * (1.0f / 1024.0f) - mean * mean;
    const float inv  = rsqrtf(var + 1e-5f);
#pragma unroll
    for (int k = 0; k < 32; ++k) {
        int c = k * 32 + lane;
        sh[c] = (vals[k] - mean) * inv * lnf_g[c] + lnf_b[c];
    }
    __syncthreads();

    for (int l = 0; l < 16; ++l) {
        int c0 = lane * 2;
        float v0 = sh[l * 64 + c0], v1 = sh[l * 64 + c0 + 1];
        float s2 = v0 + v1, ss2 = v0 * v0 + v1 * v1;
#pragma unroll
        for (int m = 16; m >= 1; m >>= 1) {
            s2  += __shfl_xor(s2,  m, 32);
            ss2 += __shfl_xor(ss2, m, 32);
        }
        float m2   = s2 * (1.0f / 64.0f);
        float var2 = ss2 * (1.0f / 64.0f) - m2 * m2;
        float inv2 = rsqrtf(var2 + 1e-5f);
        tee2d[(t * 16 + l) * 64 + c0]     = (v0 - m2) * inv2 * lnd_g[c0]     + lnd_b[c0];
        tee2d[(t * 16 + l) * 64 + c0 + 1] = (v1 - m2) * inv2 * lnd_g[c0 + 1] + lnd_b[c0 + 1];
    }
}

// ---------------------------------------------------------------------------
// Decoder on the 128 distinct rows: gelu(tee2d @ dw1 + db1) @ dw2 + db2.
// One block per (t,l).  ~10 MFLOP total: scalar fp32 FMAs, full accuracy.
// ---------------------------------------------------------------------------
__global__ __launch_bounds__(256)
void decoder_kernel(const float* __restrict__ tee2d,
                    const float* __restrict__ dw1, const float* __restrict__ db1,
                    const float* __restrict__ dw2, const float* __restrict__ db2,
                    float* __restrict__ dec) {
    __shared__ float sin_[64], sh_[128];
    const int tid = threadIdx.x, bl = blockIdx.x;   // bl = t*16 + l
    if (tid < 64) sin_[tid] = tee2d[bl * 64 + tid];
    __syncthreads();
    if (tid < 128) {
        float acc = db1[tid];
#pragma unroll 4
        for (int k = 0; k < 64; ++k) acc += sin_[k] * dw1[k * 128 + tid];
        sh_[tid] = gelu_exact(acc);
    }
    __syncthreads();
    float acc = db2[tid];
#pragma unroll 4
    for (int k = 0; k < 128; ++k) acc += sh_[k] * dw2[k * 256 + tid];
    dec[(size_t)bl * 256 + tid] = acc;
}

// ---------------------------------------------------------------------------
// out[t, i<8192, :] = dec[t*16 + (indices[t,i] >> 11), :]; zeros for i>=8192.
// NT float4 stores; one row per 64 threads (wave-uniform branch).
// ---------------------------------------------------------------------------
__global__ __launch_bounds__(256)
void gather_kernel(const float* __restrict__ dec, const int* __restrict__ indices,
                   float* __restrict__ out) {
    const int row = blockIdx.x * 4 + (threadIdx.x >> 6);
    const int c   = (threadIdx.x & 63) * 4;
    const int t   = row / NTOK;
    const int i   = row - t * NTOK;
    f4v v = (f4v){0.0f, 0.0f, 0.0f, 0.0f};
    if (i < NNODE) {
        const int l = indices[t * NNODE + i] >> 11;   // chunk = 2048
        v = *(const f4v*)(dec + ((size_t)(t * 16 + l)) * 256 + c);
    }
    __builtin_nontemporal_store(v, (f4v*)(out + (size_t)row * 256 + c));
}

// ---------------------------------------------------------------------------
extern "C" void kernel_launch(void* const* d_in, const int* in_sizes, int n_in,
                              void* d_out, int out_size, void* d_ws, size_t ws_size,
                              hipStream_t stream) {
    const float* x       = (const float*)d_in[0];
    const int*   indices = (const int*)  d_in[1];
    const float* ln1_g   = (const float*)d_in[2];
    const float* ln1_b   = (const float*)d_in[3];
    const float* w1      = (const float*)d_in[4];
    const float* b1      = (const float*)d_in[5];
    const float* w2      = (const float*)d_in[6];
    const float* b2      = (const float*)d_in[7];
    const float* lnf_g   = (const float*)d_in[8];
    const float* lnf_b   = (const float*)d_in[9];
    const float* lnd_g   = (const float*)d_in[10];
    const float* lnd_b   = (const float*)d_in[11];
    const float* dw1     = (const float*)d_in[12];
    const float* db1     = (const float*)d_in[13];
    const float* dw2     = (const float*)d_in[14];
    const float* db2     = (const float*)d_in[15];
    float* out = (float*)d_out;

    // workspace layout (256B aligned slots) -- ~9.5 MB total
    char* ws = (char*)d_ws;
    float*          comp  = (float*)         (ws + 0);        // 32 KB
    float*          tee2d = (float*)         (ws + 32768);    // 32 KB
    float*          dec   = (float*)         (ws + 65536);    // 128 KB
    __bf16*         w1p   = (__bf16*)        (ws + 196608);   // 64 KB
    __bf16*         w2p   = (__bf16*)        (ws + 262144);   // 16 KB
    int*            cnt   = (int*)           (ws + 278528);   // 8*32768*4 = 1 MB
    unsigned short* list  = (unsigned short*)(ws + 278528 + 1048576); // 8 MB

    hipMemsetAsync(comp, 0, (size_t)T_DIM * COMP_LEN * COMP_DIM * sizeof(float), stream);
    hipMemsetAsync(cnt,  0, (size_t)T_DIM * NNODES * sizeof(int), stream);

    pack_b_kernel<<<128, 256, 0, stream>>>(w1, w1p, 256, 128);
    pack_b_kernel<<<32,  256, 0, stream>>>(w2, w2p, 128, 64);

    count_fill_kernel<<<(T_DIM * NNODE) / 256, 256, 0, stream>>>(indices, cnt, list);

    encoder_kernel<<<(T_DIM * NNODES / 16) / 2, 64, 0, stream>>>(
        x, cnt, list, ln1_g, ln1_b,
        (const v16bf*)w1p, b1, (const v16bf*)w2p, b2, comp);

    comp_ln_kernel<<<T_DIM, 32, 0, stream>>>(comp, lnf_g, lnf_b, lnd_g, lnd_b, tee2d);

    decoder_kernel<<<T_DIM * COMP_LEN, 256, 0, stream>>>(tee2d, dw1, db1, dw2, db2, dec);

    gather_kernel<<<(T_DIM * NTOK) / 4, 256, 0, stream>>>(dec, indices, out);
}
